// EnhancedTransformerBlock_24163486007417
// MI455X (gfx1250) — compile-verified, matrix operand order checked by reading, not run on hardware
//
#include <hip/hip_runtime.h>
#include <hip/hip_bf16.h>

typedef __bf16 bf16;
typedef __attribute__((ext_vector_type(16))) __bf16 v16bf;
typedef __attribute__((ext_vector_type(8)))  __bf16 v8bf;
typedef __attribute__((ext_vector_type(8)))  float  v8f;

// ---------------------------------------------------------------------------
// WMMA fragment load: 16x32 bf16 operand tile, row-major source with leading
// dim `ld`. Per ISA 16-bit A layout: lanes 0-15 hold K=[0..7]+[16..23],
// lanes 16-31 hold K=[8..15]+[24..31]; B uses the same pattern on pre-
// transposed Bt[N][K]. Two b128 loads per lane.
// ---------------------------------------------------------------------------
__device__ __forceinline__ v16bf load_frag(const bf16* p, int ld, int rowbase, int k)
{
    int lane = threadIdx.x & 31;
    const bf16* r = p + (size_t)(rowbase + (lane & 15)) * (size_t)ld
                      + k + ((lane >> 4) << 3);
    v8bf lo = *(const v8bf*)(r);
    v8bf hi = *(const v8bf*)(r + 16);
    v16bf o;
#pragma unroll
    for (int i = 0; i < 8; ++i) { o[i] = lo[i]; o[i + 8] = hi[i]; }
    return o;
}

#define WMMA_BF16(a, b, c) \
    __builtin_amdgcn_wmma_f32_16x16x32_bf16(false, (a), false, (b), (short)0, (c), false, false)

// ---------------------------------------------------------------------------
// Generic bf16 WMMA GEMM:  C[M,N] = alpha * A[M,K] * Bt[N,K]^T  (+bias[n])
//                                   (+Res[m,n])   ; optional bf16 mirror.
// Batched over grid.z with (b = z/Hn, h = z%Hn) pointer strides.
// Block = 256 threads (8 waves); wave tile 32x64 (8 WMMAs / K=32 step,
// double-buffered fragment loads); block tile 256x64.
// ---------------------------------------------------------------------------
__global__ void __launch_bounds__(256)
gemm_wmma(const bf16* A, int lda, const bf16* Bt, int ldb,
          float* C, int ldc, bf16* Cbf, int K, float alpha,
          const float* bias, const float* Res, int ldr,
          long long sAb, long long sAh, long long sBb, long long sBh,
          long long sCb, long long sCh, int Hn)
{
    int z = blockIdx.z;
    int b = z / Hn, h = z % Hn;
    A  += (size_t)b * sAb + (size_t)h * sAh;
    Bt += (size_t)b * sBb + (size_t)h * sBh;
    size_t co = (size_t)b * sCb + (size_t)h * sCh;

    int wave = threadIdx.x >> 5;
    int m0 = blockIdx.y * 256 + wave * 32;
    int n0 = blockIdx.x * 64;

    v8f acc[8] = {};

    // prefetch k = 0 fragments
    v16bf a0 = load_frag(A, lda, m0,      0);
    v16bf a1 = load_frag(A, lda, m0 + 16, 0);
    v16bf b0 = load_frag(Bt, ldb, n0,      0);
    v16bf b1 = load_frag(Bt, ldb, n0 + 16, 0);
    v16bf b2 = load_frag(Bt, ldb, n0 + 32, 0);
    v16bf b3 = load_frag(Bt, ldb, n0 + 48, 0);

    for (int k = 0; k < K; k += 32) {
        v16bf ca0 = a0, ca1 = a1, cb0 = b0, cb1 = b1, cb2 = b2, cb3 = b3;
        int kn = k + 32;
        if (kn < K) {   // issue next-step loads before this step's WMMAs
            a0 = load_frag(A, lda, m0,      kn);
            a1 = load_frag(A, lda, m0 + 16, kn);
            b0 = load_frag(Bt, ldb, n0,      kn);
            b1 = load_frag(Bt, ldb, n0 + 16, kn);
            b2 = load_frag(Bt, ldb, n0 + 32, kn);
            b3 = load_frag(Bt, ldb, n0 + 48, kn);
        }
        acc[0] = WMMA_BF16(ca0, cb0, acc[0]);
        acc[1] = WMMA_BF16(ca0, cb1, acc[1]);
        acc[2] = WMMA_BF16(ca0, cb2, acc[2]);
        acc[3] = WMMA_BF16(ca0, cb3, acc[3]);
        acc[4] = WMMA_BF16(ca1, cb0, acc[4]);
        acc[5] = WMMA_BF16(ca1, cb1, acc[5]);
        acc[6] = WMMA_BF16(ca1, cb2, acc[6]);
        acc[7] = WMMA_BF16(ca1, cb3, acc[7]);
    }

    int lane  = threadIdx.x & 31;
    int nsub  = lane & 15;
    int roff  = (lane >> 4) << 3;          // lanes 16-31 -> rows +8
#pragma unroll
    for (int mi = 0; mi < 2; ++mi) {
        int rbase = m0 + mi * 16 + roff;
#pragma unroll
        for (int t = 0; t < 4; ++t) {
            int col = n0 + t * 16 + nsub;
            float bi = bias ? bias[col] : 0.f;
            v8f av = acc[mi * 4 + t];
#pragma unroll
            for (int r = 0; r < 8; ++r) {
                int row = rbase + r;
                float val = av[r] * alpha + bi;
                if (Res) val += Res[(size_t)row * ldr + col];
                size_t idx = co + (size_t)row * ldc + col;
                if (C)   C[idx]   = val;
                if (Cbf) Cbf[idx] = (bf16)val;
            }
        }
    }
}

// ---------------------------------------------------------------------------
// LayerNorm (f32 in -> bf16 out), one block per row.
// ---------------------------------------------------------------------------
__global__ void __launch_bounds__(256)
ln_bf16(const float* x, const float* g, const float* bta, bf16* out, int n)
{
    int row = blockIdx.x;
    x   += (size_t)row * n;
    out += (size_t)row * n;
    __shared__ float red[256];
    int tid = threadIdx.x;

    float s = 0.f;
    for (int i = tid; i < n; i += 256) s += x[i];
    red[tid] = s; __syncthreads();
    for (int st = 128; st > 0; st >>= 1) { if (tid < st) red[tid] += red[tid + st]; __syncthreads(); }
    float mean = red[0] / n; __syncthreads();

    float s2 = 0.f;
    for (int i = tid; i < n; i += 256) { float d = x[i] - mean; s2 += d * d; }
    red[tid] = s2; __syncthreads();
    for (int st = 128; st > 0; st >>= 1) { if (tid < st) red[tid] += red[tid + st]; __syncthreads(); }
    float rstd = rsqrtf(red[0] / n + 1e-5f);

    for (int i = tid; i < n; i += 256)
        out[i] = (bf16)((x[i] - mean) * rstd * g[i] + bta[i]);
}

// ---------------------------------------------------------------------------
// Row softmax (f32 in -> bf16 out), one block per row.
// ---------------------------------------------------------------------------
__global__ void __launch_bounds__(256)
softmax_bf16(const float* S, bf16* P, int L)
{
    int row = blockIdx.x;
    S += (size_t)row * L;
    P += (size_t)row * L;
    __shared__ float red[256];
    int tid = threadIdx.x;

    float m = -3.0e38f;
    for (int i = tid; i < L; i += 256) m = fmaxf(m, S[i]);
    red[tid] = m; __syncthreads();
    for (int st = 128; st > 0; st >>= 1) { if (tid < st) red[tid] = fmaxf(red[tid], red[tid + st]); __syncthreads(); }
    m = red[0]; __syncthreads();

    float s = 0.f;
    for (int i = tid; i < L; i += 256) s += expf(S[i] - m);
    red[tid] = s; __syncthreads();
    for (int st = 128; st > 0; st >>= 1) { if (tid < st) red[tid] += red[tid + st]; __syncthreads(); }
    float inv = 1.f / red[0];

    for (int i = tid; i < L; i += 256) P[i] = (bf16)(expf(S[i] - m) * inv);
}

// ---------------------------------------------------------------------------
// Top-k(32) sparse attention epilogue: one block per (query row) of one (b,h).
// Scores S are a 2304x2304 f32 matrix (already scaled). Selects top 32,
// softmaxes them, gathers V rows, writes the 64-wide head output slice.
// ---------------------------------------------------------------------------
#define NI 2304
#define TOPK 32
__global__ void __launch_bounds__(256)
topk_attn(const float* S, const float* V, float* O, int b, int h)
{
    __shared__ float sc[NI];
    __shared__ float rv[256];
    __shared__ int   ri[256];
    __shared__ float wsel[TOPK];
    __shared__ int   isel[TOPK];

    int row = blockIdx.x;
    int tid = threadIdx.x;
    const float* srow = S + (size_t)row * NI;
    for (int i = tid; i < NI; i += 256) sc[i] = srow[i];
    __syncthreads();

    for (int k = 0; k < TOPK; ++k) {
        float bv = -3.0e38f; int bi = 0;
        for (int i = tid; i < NI; i += 256)
            if (sc[i] > bv) { bv = sc[i]; bi = i; }
        rv[tid] = bv; ri[tid] = bi; __syncthreads();
        for (int st = 128; st > 0; st >>= 1) {
            if (tid < st && rv[tid + st] > rv[tid]) { rv[tid] = rv[tid + st]; ri[tid] = ri[tid + st]; }
            __syncthreads();
        }
        if (tid == 0) { wsel[k] = rv[0]; isel[k] = ri[0]; sc[ri[0]] = -3.0e38f; }
        __syncthreads();
    }

    if (tid == 0) {
        float mx = wsel[0];          // first selection is the max
        float sum = 0.f;
        for (int k = 0; k < TOPK; ++k) { wsel[k] = expf(wsel[k] - mx); sum += wsel[k]; }
        float inv = 1.f / sum;
        for (int k = 0; k < TOPK; ++k) wsel[k] *= inv;
    }
    __syncthreads();

    if (tid < 64) {
        float o = 0.f;
        for (int k = 0; k < TOPK; ++k)
            o += wsel[k] * V[(size_t)(b * NI + isel[k]) * 512 + h * 64 + tid];
        O[(size_t)(b * NI + row) * 512 + h * 64 + tid] = o;
    }
}

// ---------------------------------------------------------------------------
// GEGLU: h1[M,2F] f32 -> out[M,F] bf16,  out = a * gelu_exact(g)
// ---------------------------------------------------------------------------
__global__ void __launch_bounds__(256)
geglu_bf16(const float* h1, bf16* out, long long total, int F)
{
    long long i = (long long)blockIdx.x * 256 + threadIdx.x;
    if (i >= total) return;
    long long r = i / F;
    int c = (int)(i % F);
    float a = h1[(size_t)r * 2 * F + c];
    float g = h1[(size_t)r * 2 * F + F + c];
    float ge = 0.5f * g * (1.f + erff(g * 0.70710678118654752f));
    out[i] = (bf16)(a * ge);
}

// f32 -> bf16 flat convert
__global__ void __launch_bounds__(256)
cvt_bf16(const float* in, bf16* out, long long n)
{
    long long i = (long long)blockIdx.x * 256 + threadIdx.x;
    if (i < n) out[i] = (bf16)in[i];
}

// Transpose+convert: in f32 [R,Cc] (ldin) -> out bf16 [Cc,R] (ldout), batched.
__global__ void __launch_bounds__(256)
transpose_cvt(const float* in, int ldin, int R, int Cc, bf16* out, int ldout,
              long long sIb, long long sIh, long long sOb, long long sOh, int Hn)
{
    int z = blockIdx.z;
    int b = z / Hn, h = z % Hn;
    in  += (size_t)b * sIb + (size_t)h * sIh;
    out += (size_t)b * sOb + (size_t)h * sOh;
    long long i = (long long)blockIdx.x * 256 + threadIdx.x;
    if (i >= (long long)R * Cc) return;
    int r = (int)(i / Cc), c = (int)(i % Cc);
    out[(size_t)c * ldout + r] = (bf16)in[(size_t)r * ldin + c];
}

// ---------------------------------------------------------------------------
// Host side
// ---------------------------------------------------------------------------
static inline void* bump(char*& p, size_t bytes)
{
    void* r = p;
    p += (bytes + 255) & ~(size_t)255;
    return r;
}

static void gemm(hipStream_t s, const bf16* A, int lda, const bf16* Bt, int ldb,
                 float* C, int ldc, bf16* Cbf, int M, int N, int K, float alpha,
                 const float* bias, const float* res, int ldr,
                 long long sAb = 0, long long sAh = 0, long long sBb = 0, long long sBh = 0,
                 long long sCb = 0, long long sCh = 0, int Hn = 1, int nz = 1)
{
    dim3 g(N / 64, M / 256, nz), bl(256, 1, 1);
    gemm_wmma<<<g, bl, 0, s>>>(A, lda, Bt, ldb, C, ldc, Cbf, K, alpha, bias, res, ldr,
                               sAb, sAh, sBb, sBh, sCb, sCh, Hn);
}

static void tcvt(hipStream_t s, const float* in, int ldin, int R, int Cc, bf16* out, int ldout,
                 long long sIb = 0, long long sIh = 0, long long sOb = 0, long long sOh = 0,
                 int Hn = 1, int nz = 1)
{
    long long total = (long long)R * Cc;
    dim3 g((unsigned)((total + 255) / 256), 1, nz), bl(256, 1, 1);
    transpose_cvt<<<g, bl, 0, s>>>(in, ldin, R, Cc, out, ldout, sIb, sIh, sOb, sOh, Hn);
}

extern "C" void kernel_launch(void* const* d_in, const int* in_sizes, int n_in,
                              void* d_out, int out_size, void* d_ws, size_t ws_size,
                              hipStream_t stream)
{
    (void)in_sizes; (void)n_in; (void)out_size; (void)ws_size;
    const int B = 2, Ni = 2304, D = 512, Nc = 256, Cd = 768, H = 8;
    const int Mi = B * Ni;      // 4608
    const int Mc = B * Nc;      // 512
    const float scale = 0.125f; // 1/sqrt(64)

    const float* x       = (const float*)d_in[0];
    const float* context = (const float*)d_in[1];
    const float* ctx_wq  = (const float*)d_in[3];
    const float* ctx_wk  = (const float*)d_in[4];
    const float* ctx_wv  = (const float*)d_in[5];
    const float* ctx_wo  = (const float*)d_in[6];
    const float* ctx_bo  = (const float*)d_in[7];
    const float* cn_g    = (const float*)d_in[8];
    const float* cn_b    = (const float*)d_in[9];
    const float* im_wq   = (const float*)d_in[10];
    const float* im_wk   = (const float*)d_in[11];
    const float* im_wv   = (const float*)d_in[12];
    const float* im_wo   = (const float*)d_in[13];
    const float* im_bo   = (const float*)d_in[14];
    const float* xc_wq   = (const float*)d_in[15];
    const float* xc_wk   = (const float*)d_in[16];
    const float* xc_wv   = (const float*)d_in[17];
    const float* xc_wo   = (const float*)d_in[18];
    const float* xc_bo   = (const float*)d_in[19];
    const float* ffc_w1  = (const float*)d_in[20];
    const float* ffc_b1  = (const float*)d_in[21];
    const float* ffc_w2  = (const float*)d_in[22];
    const float* ffc_b2  = (const float*)d_in[23];
    const float* ffi_w1  = (const float*)d_in[24];
    const float* ffi_b1  = (const float*)d_in[25];
    const float* ffi_w2  = (const float*)d_in[26];
    const float* ffi_b2  = (const float*)d_in[27];
    const float* n1_g = (const float*)d_in[28];
    const float* n1_b = (const float*)d_in[29];
    const float* n2_g = (const float*)d_in[30];
    const float* n2_b = (const float*)d_in[31];
    const float* n3_g = (const float*)d_in[32];
    const float* n3_b = (const float*)d_in[33];

    char* wp = (char*)d_ws;
    // --- bf16 transposed weights  Wt[N][K] ---
    bf16* Wt_cq  = (bf16*)bump(wp, (size_t)512 * 768 * 2);
    bf16* Wt_ck  = (bf16*)bump(wp, (size_t)512 * 768 * 2);
    bf16* Wt_cv  = (bf16*)bump(wp, (size_t)512 * 768 * 2);
    bf16* Wt_co  = (bf16*)bump(wp, (size_t)768 * 512 * 2);
    bf16* Wt_iq  = (bf16*)bump(wp, (size_t)512 * 512 * 2);
    bf16* Wt_ik  = (bf16*)bump(wp, (size_t)512 * 512 * 2);
    bf16* Wt_iv  = (bf16*)bump(wp, (size_t)512 * 512 * 2);
    bf16* Wt_io  = (bf16*)bump(wp, (size_t)512 * 512 * 2);
    bf16* Wt_xq  = (bf16*)bump(wp, (size_t)512 * 512 * 2);
    bf16* Wt_xk  = (bf16*)bump(wp, (size_t)512 * 768 * 2);
    bf16* Wt_xv  = (bf16*)bump(wp, (size_t)512 * 768 * 2);
    bf16* Wt_xo  = (bf16*)bump(wp, (size_t)512 * 512 * 2);
    bf16* Wt_fc1 = (bf16*)bump(wp, (size_t)6144 * 768 * 2);
    bf16* Wt_fc2 = (bf16*)bump(wp, (size_t)768 * 3072 * 2);
    bf16* Wt_fi1 = (bf16*)bump(wp, (size_t)4096 * 512 * 2);
    bf16* Wt_fi2 = (bf16*)bump(wp, (size_t)512 * 2048 * 2);
    // --- activations ---
    bf16*  cn_bf  = (bf16*)bump(wp, (size_t)Mc * Cd * 2);
    bf16*  cq_bf  = (bf16*)bump(wp, (size_t)Mc * 512 * 2);
    bf16*  ck_bf  = (bf16*)bump(wp, (size_t)Mc * 512 * 2);
    float* cv_f   = (float*)bump(wp, (size_t)Mc * 512 * 4);
    bf16*  vt_c   = (bf16*)bump(wp, (size_t)16 * 64 * 256 * 2);
    float* S_c    = (float*)bump(wp, (size_t)16 * 256 * 256 * 4);
    bf16*  P_c    = (bf16*)bump(wp, (size_t)16 * 256 * 256 * 2);
    float* O_c    = (float*)bump(wp, (size_t)Mc * 512 * 4);
    bf16*  O_c_bf = (bf16*)bump(wp, (size_t)Mc * 512 * 2);
    float* c1     = (float*)bump(wp, (size_t)Mc * Cd * 4);
    bf16*  c1_bf  = (bf16*)bump(wp, (size_t)Mc * Cd * 2);
    float* h1c    = (float*)bump(wp, (size_t)Mc * 6144 * 4);
    bf16*  gac    = (bf16*)bump(wp, (size_t)Mc * 3072 * 2);
    float* c2     = (float*)bump(wp, (size_t)Mc * Cd * 4);
    bf16*  c2_bf  = (bf16*)bump(wp, (size_t)Mc * Cd * 2);
    bf16*  xn_bf  = (bf16*)bump(wp, (size_t)Mi * D * 2);
    bf16*  iq_bf  = (bf16*)bump(wp, (size_t)Mi * 512 * 2);
    bf16*  ik_bf  = (bf16*)bump(wp, (size_t)Mi * 512 * 2);
    float* iv_f   = (float*)bump(wp, (size_t)Mi * 512 * 4);
    float* S_i    = (float*)bump(wp, (size_t)Ni * Ni * 4);       // reused per (b,h)
    float* O_i    = (float*)bump(wp, (size_t)Mi * 512 * 4);
    bf16*  O_i_bf = (bf16*)bump(wp, (size_t)Mi * 512 * 2);
    float* hbuf   = (float*)bump(wp, (size_t)Mi * D * 4);
    bf16*  hn2_bf = (bf16*)bump(wp, (size_t)Mi * D * 2);
    bf16*  xq_bf  = (bf16*)bump(wp, (size_t)Mi * 512 * 2);
    bf16*  xk_bf  = (bf16*)bump(wp, (size_t)Mc * 512 * 2);
    float* xv_f   = (float*)bump(wp, (size_t)Mc * 512 * 4);
    bf16*  vt_x   = (bf16*)bump(wp, (size_t)16 * 64 * 256 * 2);
    float* S_x1   = (float*)bump(wp, (size_t)Ni * 256 * 4);      // reused per (b,h)
    bf16*  P_x1   = (bf16*)bump(wp, (size_t)Ni * 256 * 2);
    float* O_x    = (float*)bump(wp, (size_t)Mi * 512 * 4);
    bf16*  O_x_bf = (bf16*)bump(wp, (size_t)Mi * 512 * 2);
    float* h2     = (float*)bump(wp, (size_t)Mi * D * 4);
    bf16*  hn3_bf = (bf16*)bump(wp, (size_t)Mi * D * 2);
    float* h1i    = (float*)bump(wp, (size_t)Ni * 4096 * 4);     // reused per chunk
    bf16*  gai    = (bf16*)bump(wp, (size_t)Mi * 2048 * 2);
    float* out    = (float*)d_out;

    // ---- weight transposes (f32 [K,N] -> bf16 [N,K]) ----
    tcvt(stream, ctx_wq, 512, 768, 512, Wt_cq, 768);
    tcvt(stream, ctx_wk, 512, 768, 512, Wt_ck, 768);
    tcvt(stream, ctx_wv, 512, 768, 512, Wt_cv, 768);
    tcvt(stream, ctx_wo, 768, 512, 768, Wt_co, 512);
    tcvt(stream, im_wq, 512, 512, 512, Wt_iq, 512);
    tcvt(stream, im_wk, 512, 512, 512, Wt_ik, 512);
    tcvt(stream, im_wv, 512, 512, 512, Wt_iv, 512);
    tcvt(stream, im_wo, 512, 512, 512, Wt_io, 512);
    tcvt(stream, xc_wq, 512, 512, 512, Wt_xq, 512);
    tcvt(stream, xc_wk, 512, 768, 512, Wt_xk, 768);
    tcvt(stream, xc_wv, 512, 768, 512, Wt_xv, 768);
    tcvt(stream, xc_wo, 512, 512, 512, Wt_xo, 512);
    tcvt(stream, ffc_w1, 6144, 768, 6144, Wt_fc1, 768);
    tcvt(stream, ffc_w2, 768, 3072, 768, Wt_fc2, 3072);
    tcvt(stream, ffi_w1, 4096, 512, 4096, Wt_fi1, 512);
    tcvt(stream, ffi_w2, 512, 2048, 512, Wt_fi2, 2048);

    // ================= context branch =================
    ln_bf16<<<Mc, 256, 0, stream>>>(context, cn_g, cn_b, cn_bf, Cd);
    gemm(stream, cn_bf, Cd, Wt_cq, Cd, nullptr, 512, cq_bf, Mc, 512, Cd, 1.f, nullptr, nullptr, 0);
    gemm(stream, cn_bf, Cd, Wt_ck, Cd, nullptr, 512, ck_bf, Mc, 512, Cd, 1.f, nullptr, nullptr, 0);
    gemm(stream, cn_bf, Cd, Wt_cv, Cd, cv_f, 512, nullptr, Mc, 512, Cd, 1.f, nullptr, nullptr, 0);
    // per-head V^T : [z][64][256]
    tcvt(stream, cv_f, 512, 256, 64, vt_c, 256,
         (long long)256 * 512, 64, (long long)8 * 16384, 16384, H, 16);
    // S = scale * Q K^T  (batched over 16 (b,h))
    gemm(stream, cq_bf, 512, ck_bf, 512, S_c, 256, nullptr, 256, 256, 64, scale,
         nullptr, nullptr, 0, (long long)256 * 512, 64, (long long)256 * 512, 64,
         (long long)8 * 65536, 65536, H, 16);
    softmax_bf16<<<16 * 256, 256, 0, stream>>>(S_c, P_c, 256);
    // O = P V
    gemm(stream, P_c, 256, vt_c, 256, O_c, 512, O_c_bf, 256, 64, 256, 1.f,
         nullptr, nullptr, 0, (long long)8 * 65536, 65536, (long long)8 * 16384, 16384,
         (long long)256 * 512, 64, H, 16);
    // c1 = context + O @ wo + bo
    gemm(stream, O_c_bf, 512, Wt_co, 512, c1, Cd, c1_bf, Mc, Cd, 512, 1.f, ctx_bo, context, Cd);
    // context GEGLU FF
    gemm(stream, c1_bf, Cd, Wt_fc1, Cd, h1c, 6144, nullptr, Mc, 6144, Cd, 1.f, ffc_b1, nullptr, 0);
    {
        long long tot = (long long)Mc * 3072;
        geglu_bf16<<<(unsigned)((tot + 255) / 256), 256, 0, stream>>>(h1c, gac, tot, 3072);
    }
    gemm(stream, gac, 3072, Wt_fc2, 3072, c2, Cd, c2_bf, Mc, Cd, 3072, 1.f, ffc_b2, c1, Cd);

    // ================= image top-k self-attention =================
    ln_bf16<<<Mi, 256, 0, stream>>>(x, n1_g, n1_b, xn_bf, D);
    gemm(stream, xn_bf, D, Wt_iq, D, nullptr, 512, iq_bf, Mi, 512, D, 1.f, nullptr, nullptr, 0);
    gemm(stream, xn_bf, D, Wt_ik, D, nullptr, 512, ik_bf, Mi, 512, D, 1.f, nullptr, nullptr, 0);
    gemm(stream, xn_bf, D, Wt_iv, D, iv_f, 512, nullptr, Mi, 512, D, 1.f, nullptr, nullptr, 0);
    for (int z = 0; z < 16; ++z) {
        int b = z / H, hh = z % H;
        const bf16* Aq = iq_bf + (size_t)b * Ni * 512 + hh * 64;
        const bf16* Bk = ik_bf + (size_t)b * Ni * 512 + hh * 64;
        gemm(stream, Aq, 512, Bk, 512, S_i, Ni, nullptr, Ni, Ni, 64, scale, nullptr, nullptr, 0);
        topk_attn<<<Ni, 256, 0, stream>>>(S_i, iv_f, O_i, b, hh);
    }
    {
        long long tot = (long long)Mi * 512;
        cvt_bf16<<<(unsigned)((tot + 255) / 256), 256, 0, stream>>>(O_i, O_i_bf, tot);
    }
    // h = x + O @ im_wo + im_bo
    gemm(stream, O_i_bf, 512, Wt_io, 512, hbuf, D, nullptr, Mi, D, 512, 1.f, im_bo, x, D);

    // ================= image -> context cross-attention =================
    ln_bf16<<<Mi, 256, 0, stream>>>(hbuf, n2_g, n2_b, hn2_bf, D);
    gemm(stream, hn2_bf, D, Wt_xq, D, nullptr, 512, xq_bf, Mi, 512, D, 1.f, nullptr, nullptr, 0);
    gemm(stream, c2_bf, Cd, Wt_xk, Cd, nullptr, 512, xk_bf, Mc, 512, Cd, 1.f, nullptr, nullptr, 0);
    gemm(stream, c2_bf, Cd, Wt_xv, Cd, xv_f, 512, nullptr, Mc, 512, Cd, 1.f, nullptr, nullptr, 0);
    tcvt(stream, xv_f, 512, 256, 64, vt_x, 256,
         (long long)256 * 512, 64, (long long)8 * 16384, 16384, H, 16);
    for (int z = 0; z < 16; ++z) {
        int b = z / H, hh = z % H;
        const bf16* Aq = xq_bf + (size_t)b * Ni * 512 + hh * 64;
        const bf16* Bk = xk_bf + (size_t)b * Nc * 512 + hh * 64;
        gemm(stream, Aq, 512, Bk, 512, S_x1, 256, nullptr, Ni, 256, 64, scale, nullptr, nullptr, 0);
        softmax_bf16<<<Ni, 256, 0, stream>>>(S_x1, P_x1, 256);
        float* Oz = O_x + (size_t)b * Ni * 512 + hh * 64;
        bf16*  Ozb = O_x_bf + (size_t)b * Ni * 512 + hh * 64;
        gemm(stream, P_x1, 256, vt_x + (size_t)z * 16384, 256, Oz, 512, Ozb,
             Ni, 64, 256, 1.f, nullptr, nullptr, 0);
    }
    // h2 = h + O_x @ xc_wo + xc_bo
    gemm(stream, O_x_bf, 512, Wt_xo, 512, h2, D, nullptr, Mi, D, 512, 1.f, xc_bo, hbuf, D);

    // ================= image GEGLU FF =================
    ln_bf16<<<Mi, 256, 0, stream>>>(h2, n3_g, n3_b, hn3_bf, D);
    for (int chunk = 0; chunk < 2; ++chunk) {
        const bf16* Ah = hn3_bf + (size_t)chunk * Ni * D;
        gemm(stream, Ah, D, Wt_fi1, D, h1i, 4096, nullptr, Ni, 4096, D, 1.f, ffi_b1, nullptr, 0);
        long long tot = (long long)Ni * 2048;
        geglu_bf16<<<(unsigned)((tot + 255) / 256), 256, 0, stream>>>(
            h1i, gai + (size_t)chunk * Ni * 2048, tot, 2048);
    }
    // out = h2 + ga @ ffi_w2 + ffi_b2
    gemm(stream, gai, 2048, Wt_fi2, 2048, out, D, nullptr, Mi, D, 2048, 1.f, ffi_b2, h2, D);
}
